// GAT_19318762897898
// MI455X (gfx1250) — matrix operand order repaired and checked by reading, not compiled, not run on hardware
//
#include <hip/hip_runtime.h>

typedef __attribute__((ext_vector_type(16))) _Float16 v16h;
typedef __attribute__((ext_vector_type(8)))  _Float16 v8h;
typedef __attribute__((ext_vector_type(8)))  float    v8f;

constexpr int N_NODES = 50000;
constexpr int IN_F    = 32;
constexpr int H_F     = 32;
constexpr int OUT_F   = 16;
constexpr int HEADS   = 8;
constexpr int N_EDGES = 640000;
constexpr int ET      = N_EDGES + N_NODES;   // with self loops
constexpr int NO1     = HEADS * H_F;         // 256
constexpr int NO2     = HEADS * OUT_F;       // 128
constexpr float NEG_SLOPE = 0.2f;
constexpr float EPS = 1e-16f;

static inline unsigned cdiv(unsigned a, unsigned b) { return (a + b - 1) / b; }

// ---------- elementwise helpers ----------
__global__ void k_f32_to_f16(const float* __restrict__ in, _Float16* __restrict__ out, int n) {
    int i = blockIdx.x * blockDim.x + threadIdx.x;
    if (i < n) out[i] = (_Float16)in[i];
}

__global__ void k_fill_u32(unsigned* __restrict__ p, unsigned v, int n) {
    int i = blockIdx.x * blockDim.x + threadIdx.x;
    if (i < n) p[i] = v;
}

// ---------- pack weights f32 [32, Nw] -> f16 in per-lane WMMA-B register order ----------
// Packed layout: tile nt (16 cols), lane l (0..31), j (0..15):
//   Bp[(nt*32 + l)*16 + j] = (f16) W[(kbb + j)*Nw + nt*16 + (l&15)],  kbb = (l&16)?16:0
// so each GEMM wave reads its whole B fragment as one contiguous 32B chunk.
__global__ void k_pack_b(const float* __restrict__ W, _Float16* __restrict__ Bp, int Nw) {
    int t = blockIdx.x * blockDim.x + threadIdx.x;
    if (t >= Nw * 32) return;
    int j   = t & 15;
    int l   = (t >> 4) & 31;
    int nt  = t >> 9;                      // 512 packed elements per 16-col tile
    int kbb = (l & 16) ? 16 : 0;
    int col = (nt << 4) + (l & 15);
    Bp[t] = (_Float16)W[(size_t)(kbb + j) * Nw + col];
}

// ---------- WMMA GEMM: C[M,Nw] = A[M,32](f16) * Bp(packed f16), K = 32 fixed ----------
// One wave computes one 16x16 tile with a single v_wmma_f32_16x16x32_f16.
__global__ void k_gemm_wmma(const _Float16* __restrict__ A, const _Float16* __restrict__ Bp,
                            float* __restrict__ C, int Mtiles, int Nw) {
    int wid  = (int)((blockIdx.x * blockDim.x + threadIdx.x) >> 5);
    int lane = threadIdx.x & 31;
    int Ntiles = Nw >> 4;
    if (wid >= Mtiles * Ntiles) return;
    int mt = wid / Ntiles, nt = wid - mt * Ntiles;
    int m0 = mt << 4, n0 = nt << 4;

    // A tile: 16x32 f16. lanes 0-15 -> K {0..15}, lanes 16-31 -> K {16..31}
    int row = m0 + (lane & 15);
    int kb  = (lane & 16) ? 16 : 0;
    const _Float16* ap = A + (size_t)row * 32 + kb;
    v16h av = *(const v16h*)(ap);

    // B fragment: one contiguous 32B load from the pre-swizzled weight buffer
    v16h bv = *(const v16h*)(Bp + ((size_t)nt * 32 + lane) * 16);

    v8f c = {0.f, 0.f, 0.f, 0.f, 0.f, 0.f, 0.f, 0.f};
    v8f d = __builtin_amdgcn_wmma_f32_16x16x32_f16(false, av, false, bv, (short)0, c, false, false);

    // C/D: VGPR r -> row m0 + (lane>=16 ? 8 : 0) + r, col = n0 + (lane&15)
    int crow = m0 + ((lane & 16) ? 8 : 0);
    float* cp = C + (size_t)crow * Nw + n0 + (lane & 15);
#pragma unroll
    for (int r = 0; r < 8; ++r) cp[(size_t)r * Nw] = d[r];
}

// ---------- per-(node,head) attention halves ----------
__global__ void k_attn(const float* __restrict__ h, const float* __restrict__ a_s,
                       const float* __restrict__ a_d, float* __restrict__ os,
                       float* __restrict__ od, int n_nodes, int C) {
    int t = blockIdx.x * blockDim.x + threadIdx.x;
    if (t >= n_nodes * HEADS) return;
    int hh = t & (HEADS - 1);
    const float* hp = h + (size_t)t * C;   // t = n*HEADS+hh, h is [N][HEADS][C]
    const float* sp = a_s + hh * C;
    const float* dp = a_d + hh * C;
    float ss = 0.f, sd = 0.f;
    for (int c = 0; c < C; ++c) { float v = hp[c]; ss += v * sp[c]; sd += v * dp[c]; }
    os[t] = ss; od[t] = sd;
}

// ---------- ordered-int float max helpers ----------
__device__ __forceinline__ int f2ord(float f) {
    int i = __float_as_int(f);
    return i >= 0 ? i : (i ^ 0x7fffffff);
}
__device__ __forceinline__ float ord2f(int i) {
    return __int_as_float(i >= 0 ? i : (i ^ 0x7fffffff));
}

__device__ __forceinline__ void edge_sd(int e, const long long* ei, int& s, int& d) {
    if (e < N_EDGES) { s = (int)ei[e]; d = (int)ei[N_EDGES + e]; }
    else             { s = d = e - N_EDGES; }   // implicit self loops
}

// ---------- edge pass 1: segment max of leaky-relu logits ----------
__global__ void k_edge_max(const long long* __restrict__ ei, const float* __restrict__ as_,
                           const float* __restrict__ ad_, int* __restrict__ mI) {
    int t = blockIdx.x * blockDim.x + threadIdx.x;
    if (t >= ET * HEADS) return;
    int e = t >> 3, hh = t & 7, s, d;
    edge_sd(e, ei, s, d);
    float v = as_[s * HEADS + hh] + ad_[d * HEADS + hh];
    v = v > 0.f ? v : NEG_SLOPE * v;
    atomicMax(&mI[d * HEADS + hh], f2ord(v));
}

// ---------- edge pass 2: segment sum of exp ----------
__global__ void k_edge_sum(const long long* __restrict__ ei, const float* __restrict__ as_,
                           const float* __restrict__ ad_, const int* __restrict__ mI,
                           float* __restrict__ den) {
    int t = blockIdx.x * blockDim.x + threadIdx.x;
    if (t >= ET * HEADS) return;
    int e = t >> 3, hh = t & 7, s, d;
    edge_sd(e, ei, s, d);
    float v = as_[s * HEADS + hh] + ad_[d * HEADS + hh];
    v = v > 0.f ? v : NEG_SLOPE * v;
    float m = ord2f(mI[d * HEADS + hh]);
    atomicAdd(&den[d * HEADS + hh], expf(v - m));
}

// ---------- edge pass 3: weighted aggregate, fused mean over heads ----------
__global__ void k_edge_agg(const long long* __restrict__ ei, const float* __restrict__ as_,
                           const float* __restrict__ ad_, const int* __restrict__ mI,
                           const float* __restrict__ den, const float* __restrict__ hsrc,
                           float* __restrict__ out, int C) {
    int t = blockIdx.x * blockDim.x + threadIdx.x;
    if (t >= ET * HEADS) return;
    int e = t >> 3, hh = t & 7, s, d;
    edge_sd(e, ei, s, d);
    float v = as_[s * HEADS + hh] + ad_[d * HEADS + hh];
    v = v > 0.f ? v : NEG_SLOPE * v;
    float m  = ord2f(mI[d * HEADS + hh]);
    float ex = expf(v - m);
    float alpha = ex / (den[d * HEADS + hh] + EPS);
    float scale = alpha * (1.0f / HEADS);          // fused head-mean
    const float* hp = hsrc + ((size_t)s * HEADS + hh) * C;
    float* op = out + (size_t)d * C;
    for (int c = 0; c < C; ++c) atomicAdd(&op[c], scale * hp[c]);
}

// ---------- bias + relu -> f16 (feeds layer-2 WMMA) ----------
__global__ void k_relu_bias_h(const float* __restrict__ in, const float* __restrict__ b,
                              _Float16* __restrict__ out, int n, int C) {
    int i = blockIdx.x * blockDim.x + threadIdx.x;
    if (i < n) {
        float v = in[i] + b[i & (C - 1)];
        out[i] = (_Float16)(v > 0.f ? v : 0.f);
    }
}

// ---------- bias + log_softmax over 16 classes ----------
__global__ void k_logsoftmax(const float* __restrict__ in, const float* __restrict__ b,
                             float* __restrict__ out, int n_nodes) {
    int nId = blockIdx.x * blockDim.x + threadIdx.x;
    if (nId >= n_nodes) return;
    float v[OUT_F];
    float m = -3.4e38f;
#pragma unroll
    for (int c = 0; c < OUT_F; ++c) {
        v[c] = in[(size_t)nId * OUT_F + c] + b[c];
        m = v[c] > m ? v[c] : m;
    }
    float s = 0.f;
#pragma unroll
    for (int c = 0; c < OUT_F; ++c) s += expf(v[c] - m);
    float lse = m + logf(s);
#pragma unroll
    for (int c = 0; c < OUT_F; ++c) out[(size_t)nId * OUT_F + c] = v[c] - lse;
}

extern "C" void kernel_launch(void* const* d_in, const int* in_sizes, int n_in,
                              void* d_out, int out_size, void* d_ws, size_t ws_size,
                              hipStream_t stream) {
    const float*     x    = (const float*)d_in[0];
    const long long* ei   = (const long long*)d_in[1];
    const float*     W1   = (const float*)d_in[2];
    const float*     aS1  = (const float*)d_in[3];
    const float*     aD1  = (const float*)d_in[4];
    const float*     b1   = (const float*)d_in[5];
    const float*     W2   = (const float*)d_in[6];
    const float*     aS2  = (const float*)d_in[7];
    const float*     aD2  = (const float*)d_in[8];
    const float*     b2   = (const float*)d_in[9];
    float* out = (float*)d_out;

    // ---- workspace layout (256B aligned) ----
    char* ws = (char*)d_ws;
    size_t o = 0;
    auto alloc = [&](size_t bytes) { size_t r = o; o = (o + bytes + 255) & ~(size_t)255; return r; };
    _Float16* xh   = (_Float16*)(ws + alloc((size_t)N_NODES * IN_F * 2));  // x f16 / reused for relu(h) f16
    _Float16* bp1  = (_Float16*)(ws + alloc((size_t)IN_F * NO1 * 2));      // W1 packed for WMMA
    _Float16* bp2  = (_Float16*)(ws + alloc((size_t)H_F * NO2 * 2));       // W2 packed for WMMA
    float*    h1   = (float*)   (ws + alloc((size_t)N_NODES * NO1 * 4));   // layer1 proj / reused as layer2 proj
    float*    as_  = (float*)   (ws + alloc((size_t)N_NODES * HEADS * 4));
    float*    ad_  = (float*)   (ws + alloc((size_t)N_NODES * HEADS * 4));
    int*      mI   = (int*)     (ws + alloc((size_t)N_NODES * HEADS * 4));
    float*    den  = (float*)   (ws + alloc((size_t)N_NODES * HEADS * 4));
    float*    hmid = (float*)   (ws + alloc((size_t)N_NODES * H_F * 4));   // layer1 aggregate (head-mean)
    float*    agg2 = (float*)   (ws + alloc((size_t)N_NODES * OUT_F * 4)); // layer2 aggregate

    const unsigned TB  = 256;
    const unsigned nNH = N_NODES * HEADS;
    const unsigned nEH = ET * HEADS;

    // ---- converts / weight packing ----
    k_f32_to_f16<<<cdiv(N_NODES * IN_F, TB), TB, 0, stream>>>(x, xh, N_NODES * IN_F);
    k_pack_b<<<cdiv(NO1 * 32, TB), TB, 0, stream>>>(W1, bp1, NO1);
    k_pack_b<<<cdiv(NO2 * 32, TB), TB, 0, stream>>>(W2, bp2, NO2);

    // ================= Layer 1 =================
    {
        unsigned waves = (N_NODES / 16) * (NO1 / 16);
        k_gemm_wmma<<<cdiv(waves * 32, TB), TB, 0, stream>>>(xh, bp1, h1, N_NODES / 16, NO1);
    }
    k_attn<<<cdiv(nNH, TB), TB, 0, stream>>>(h1, aS1, aD1, as_, ad_, N_NODES, H_F);
    k_fill_u32<<<cdiv(nNH, TB), TB, 0, stream>>>((unsigned*)mI, 0x80000000u, nNH);
    k_fill_u32<<<cdiv(nNH, TB), TB, 0, stream>>>((unsigned*)den, 0u, nNH);
    k_fill_u32<<<cdiv(N_NODES * H_F, TB), TB, 0, stream>>>((unsigned*)hmid, 0u, N_NODES * H_F);
    k_edge_max<<<cdiv(nEH, TB), TB, 0, stream>>>(ei, as_, ad_, mI);
    k_edge_sum<<<cdiv(nEH, TB), TB, 0, stream>>>(ei, as_, ad_, mI, den);
    k_edge_agg<<<cdiv(nEH, TB), TB, 0, stream>>>(ei, as_, ad_, mI, den, h1, hmid, H_F);
    k_relu_bias_h<<<cdiv(N_NODES * H_F, TB), TB, 0, stream>>>(hmid, b1, xh, N_NODES * H_F, H_F);

    // ================= Layer 2 =================
    {
        unsigned waves = (N_NODES / 16) * (NO2 / 16);
        k_gemm_wmma<<<cdiv(waves * 32, TB), TB, 0, stream>>>(xh, bp2, h1, N_NODES / 16, NO2);
    }
    k_attn<<<cdiv(nNH, TB), TB, 0, stream>>>(h1, aS2, aD2, as_, ad_, N_NODES, OUT_F);
    k_fill_u32<<<cdiv(nNH, TB), TB, 0, stream>>>((unsigned*)mI, 0x80000000u, nNH);
    k_fill_u32<<<cdiv(nNH, TB), TB, 0, stream>>>((unsigned*)den, 0u, nNH);
    k_fill_u32<<<cdiv(N_NODES * OUT_F, TB), TB, 0, stream>>>((unsigned*)agg2, 0u, N_NODES * OUT_F);
    k_edge_max<<<cdiv(nEH, TB), TB, 0, stream>>>(ei, as_, ad_, mI);
    k_edge_sum<<<cdiv(nEH, TB), TB, 0, stream>>>(ei, as_, ad_, mI, den);
    k_edge_agg<<<cdiv(nEH, TB), TB, 0, stream>>>(ei, as_, ad_, mI, den, h1, agg2, OUT_F);

    k_logsoftmax<<<cdiv(N_NODES, TB), TB, 0, stream>>>(agg2, b2, out, N_NODES);
    (void)in_sizes; (void)n_in; (void)out_size; (void)ws_size;
}
